// PointTrans_Layer_86071144612518
// MI455X (gfx1250) — matrix operand order repaired and checked by reading, not compiled, not run on hardware
//
#include <hip/hip_runtime.h>

// ---------------------------------------------------------------------------
// CDNA5 (gfx1250) PointTransformerConv: bf16 WMMA for all GEMM-shaped work,
// L2 atomics for the segment softmax / scatter-add, NT hints on edge streams.
// ---------------------------------------------------------------------------

typedef __attribute__((ext_vector_type(8)))  float          v8f;
typedef __attribute__((ext_vector_type(8)))  unsigned short v8u;
typedef __attribute__((ext_vector_type(16))) unsigned short v16u;
typedef __attribute__((ext_vector_type(16))) __bf16         v16bf;

// float -> bf16 bits, round-to-nearest-even
__device__ __forceinline__ unsigned short f2bf(float f) {
  unsigned int b = __builtin_bit_cast(unsigned int, f);
  b += 0x7fffu + ((b >> 16) & 1u);
  return (unsigned short)(b >> 16);
}

// order-preserving f32 <-> u32 encoding (for atomicMax-based segment max)
__device__ __forceinline__ unsigned int fenc(float f) {
  unsigned int b = __builtin_bit_cast(unsigned int, f);
  return (b & 0x80000000u) ? ~b : (b | 0x80000000u);
}
__device__ __forceinline__ float fdec(unsigned int u) {
  unsigned int b = (u & 0x80000000u) ? (u & 0x7fffffffu) : ~u;
  return __builtin_bit_cast(float, b);
}

__device__ __forceinline__ v16u catu(v8u lo, v8u hi) {
  return __builtin_shufflevector(lo, hi, 0,1,2,3,4,5,6,7,8,9,10,11,12,13,14,15);
}

__device__ __forceinline__ v8f wmma_bf16(v16u a, v16u b, v8f c) {
  return __builtin_amdgcn_wmma_f32_16x16x32_bf16(
      false, __builtin_bit_cast(v16bf, a),
      false, __builtin_bit_cast(v16bf, b),
      (short)0, c, false, false);
}

// A-fragment (16x32 bf16), contiguous k order (loadB uses the same):
//   lane half h = lane/16, row M = lane%16
//   elements 0..7  = K = 32*kt + 16*h + (0..7)     (contiguous 16B)
//   elements 8..15 = K = 32*kt + 16*h + 8 + (0..7) (contiguous 16B)
__device__ __forceinline__ v16u loadA(const unsigned short* X, int row, int kt, int lane) {
  int half = lane >> 4;
  const unsigned short* p = X + row * 64 + kt * 32 + half * 16;
  v8u lo = *(const v8u*)p;
  v8u hi = *(const v8u*)(p + 8);
  return catu(lo, hi);
}

// B-fragment (32x16 bf16): lane n = lane%16 (+16*nt), elements e -> K = 32*kt + 16*(lane/16) + e.
// Weights stored TRANSPOSED (WT[n][k], 64x64) so this is one contiguous 32B load.
__device__ __forceinline__ v16u loadB(const unsigned short* WT, int nt, int kt, int lane) {
  int n  = nt * 16 + (lane & 15);
  int k0 = kt * 32 + ((lane >> 4) << 4);
  return *(const v16u*)(WT + n * 64 + k0);
}

// ---------------------------------------------------------------------------
// Kernel 0: pack x -> bf16, transpose+pack six 64x64 weights, init m/s/out
// ---------------------------------------------------------------------------
__global__ __launch_bounds__(256) void k_init(
    const float* __restrict__ x,
    const float* __restrict__ Wlin, const float* __restrict__ Wsrc, const float* __restrict__ Wdst,
    const float* __restrict__ posW2, const float* __restrict__ attnW1, const float* __restrict__ attnW2,
    unsigned short* __restrict__ xb, unsigned short* __restrict__ wbuf,
    unsigned int* __restrict__ m_enc, float* __restrict__ s, float* __restrict__ out, int NC) {
  int idx = blockIdx.x * 256 + threadIdx.x;
  if (idx < NC) {
    xb[idx]    = f2bf(x[idx]);
    m_enc[idx] = 0x007fffffu;   // fenc(-inf)
    s[idx]     = 0.0f;
    out[idx]   = 0.0f;
  }
  if (idx < 4096) {
    int k = idx >> 6, n = idx & 63;
    int t = n * 64 + k;         // transposed position
    wbuf[0 * 4096 + t] = f2bf(Wlin[idx]);
    wbuf[1 * 4096 + t] = f2bf(Wsrc[idx]);
    wbuf[2 * 4096 + t] = f2bf(Wdst[idx]);
    wbuf[3 * 4096 + t] = f2bf(posW2[idx]);
    wbuf[4 * 4096 + t] = f2bf(attnW1[idx]);
    wbuf[5 * 4096 + t] = f2bf(attnW2[idx]);
  }
}

// ---------------------------------------------------------------------------
// Kernel 1: node GEMMs  x_lin/a_src/a_dst = x @ {Wlin,Wsrc,Wdst}
// one wave per (16-row tile, matrix); 8 WMMAs per wave
// ---------------------------------------------------------------------------
__global__ __launch_bounds__(256) void k_node_gemm(
    const unsigned short* __restrict__ xb, const unsigned short* __restrict__ wbuf,
    float* __restrict__ xlin, float* __restrict__ asrc, float* __restrict__ adst, int N) {
  int wave = threadIdx.x >> 5, lane = threadIdx.x & 31;
  int unit = blockIdx.x * 8 + wave;
  int mtiles = (N + 15) >> 4;
  if (unit >= mtiles * 3) return;
  int mt = unit / 3, mat = unit - mt * 3;
  const unsigned short* WT = wbuf + mat * 4096;
  float* out = (mat == 0) ? xlin : (mat == 1) ? asrc : adst;

  int M = lane & 15, half = lane >> 4;
  bool full = (mt * 16 + 16 <= N);           // wave-uniform
  int row = mt * 16 + M; if (row >= N) row = N - 1;
  v16u a0 = loadA(xb, row, 0, lane);
  v16u a1 = loadA(xb, row, 1, lane);

#pragma unroll
  for (int nt = 0; nt < 4; ++nt) {
    v8f c = {};
    c = wmma_bf16(a0, loadB(WT, nt, 0, lane), c);
    c = wmma_bf16(a1, loadB(WT, nt, 1, lane), c);
    int n = nt * 16 + M;
    float* op = out + (size_t)(mt * 16 + 8 * half) * 64 + n;
    if (full) {
#pragma unroll
      for (int v = 0; v < 8; ++v) op[v * 64] = c[v];
    } else {
#pragma unroll
      for (int v = 0; v < 8; ++v) {
        int r = mt * 16 + v + 8 * half;
        if (r < N) out[r * 64 + n] = c[v];
      }
    }
  }
}

// ---------------------------------------------------------------------------
// Kernel 2: per-edge MLPs.  One wave owns a 16-edge tile; 24 WMMAs per tile.
//   delta = pos_nn(pos[dst]-pos[src]) ; alpha = attn_nn(a_dst[dst]-a_src[src]+delta)
// Wave-private LDS staging re-layouts between WMMA C-layout and A-fragments.
// ---------------------------------------------------------------------------
#define EW 8
__global__ __launch_bounds__(256) void k_edge_mlp(
    const float* __restrict__ pos,
    const int* __restrict__ src, const int* __restrict__ dst,
    const float* __restrict__ asrc, const float* __restrict__ adst,
    const unsigned short* __restrict__ wbuf,
    const float* __restrict__ posW1, const float* __restrict__ posb1, const float* __restrict__ posb2,
    const float* __restrict__ attnb1, const float* __restrict__ attnb2,
    float* __restrict__ delta, float* __restrict__ alpha, int E) {
  __shared__ __align__(16) unsigned short sbuf[EW][1024];  // bf16 A-stage [16][64]
  __shared__ float hbuf[EW][1024];                         // f32 [16][64]
  __shared__ float dp[EW][64];                             // [16][4] pos deltas
  __shared__ int   gsi[EW][16];
  __shared__ int   gdi[EW][16];

  int wave = threadIdx.x >> 5, lane = threadIdx.x & 31;
  int tile = blockIdx.x * EW + wave;
  int ebase = tile * 16;
  if (ebase >= E) return;
  int M = lane & 15, half = lane >> 4;
  bool full = (ebase + 16 <= E);             // wave-uniform

  const unsigned short* posW2T  = wbuf + 3 * 4096;
  const unsigned short* attnW1T = wbuf + 4 * 4096;
  const unsigned short* attnW2T = wbuf + 5 * 4096;

  // gather edge endpoints + pos deltas (wave-internal LDS; DS ops are in-order)
  if (lane < 16) {
    int e = ebase + lane; if (e >= E) e = E - 1;
    int sj = src[e], di = dst[e];
    gsi[wave][lane] = sj; gdi[wave][lane] = di;
    dp[wave][lane * 4 + 0] = pos[di * 3 + 0] - pos[sj * 3 + 0];
    dp[wave][lane * 4 + 1] = pos[di * 3 + 1] - pos[sj * 3 + 1];
    dp[wave][lane * 4 + 2] = pos[di * 3 + 2] - pos[sj * 3 + 2];
  }

  // pos_nn layer 1 (K=3, VALU) -> relu -> bf16 stage
  for (int idx = lane; idx < 1024; idx += 32) {
    int m = idx >> 6, n = idx & 63;
    float acc = posb1[n]
              + dp[wave][m * 4 + 0] * posW1[n]
              + dp[wave][m * 4 + 1] * posW1[64 + n]
              + dp[wave][m * 4 + 2] * posW1[128 + n];
    sbuf[wave][idx] = f2bf(acc > 0.f ? acc : 0.f);
  }

  v16u a0 = loadA(&sbuf[wave][0], M, 0, lane);
  v16u a1 = loadA(&sbuf[wave][0], M, 1, lane);

  // pos_nn layer 2 (WMMA): delta -> hbuf (f32 [m][n]) and global (NT stream)
#pragma unroll
  for (int nt = 0; nt < 4; ++nt) {
    v8f c = {};
    c = wmma_bf16(a0, loadB(posW2T, nt, 0, lane), c);
    c = wmma_bf16(a1, loadB(posW2T, nt, 1, lane), c);
    int n = nt * 16 + M;
    float bias = posb2[n];
    float* dpnt = delta + (size_t)(ebase + 8 * half) * 64 + n;
    if (full) {
#pragma unroll
      for (int v = 0; v < 8; ++v) {
        float val = c[v] + bias;
        hbuf[wave][(v + 8 * half) * 64 + n] = val;
        __builtin_nontemporal_store(val, dpnt + v * 64);
      }
    } else {
#pragma unroll
      for (int v = 0; v < 8; ++v) {
        float val = c[v] + bias;
        hbuf[wave][(v + 8 * half) * 64 + n] = val;
        int e = ebase + v + 8 * half;
        if (e < E) delta[e * 64 + n] = val;
      }
    }
  }

  // h = a_dst[dst] - a_src[src] + delta  -> bf16 stage
  for (int idx = lane; idx < 1024; idx += 32) {
    int m = idx >> 6, n = idx & 63;
    float v = adst[gdi[wave][m] * 64 + n] - asrc[gsi[wave][m] * 64 + n] + hbuf[wave][idx];
    sbuf[wave][idx] = f2bf(v);
  }

  // attn_nn layer 1 (WMMA) -> relu
  a0 = loadA(&sbuf[wave][0], M, 0, lane);
  a1 = loadA(&sbuf[wave][0], M, 1, lane);
  v8f u[4];
#pragma unroll
  for (int nt = 0; nt < 4; ++nt) {
    v8f c = {};
    c = wmma_bf16(a0, loadB(attnW1T, nt, 0, lane), c);
    c = wmma_bf16(a1, loadB(attnW1T, nt, 1, lane), c);
    int n = nt * 16 + M;
    float bias = attnb1[n];
#pragma unroll
    for (int v = 0; v < 8; ++v) {
      float t = c[v] + bias;
      c[v] = t > 0.f ? t : 0.f;
    }
    u[nt] = c;
  }
  // restage u as bf16 A-layout
#pragma unroll
  for (int nt = 0; nt < 4; ++nt) {
    int n = nt * 16 + M;
#pragma unroll
    for (int v = 0; v < 8; ++v)
      sbuf[wave][(v + 8 * half) * 64 + n] = f2bf(u[nt][v]);
  }

  // attn_nn layer 2 (WMMA) -> alpha (NT stream)
  a0 = loadA(&sbuf[wave][0], M, 0, lane);
  a1 = loadA(&sbuf[wave][0], M, 1, lane);
#pragma unroll
  for (int nt = 0; nt < 4; ++nt) {
    v8f c = {};
    c = wmma_bf16(a0, loadB(attnW2T, nt, 0, lane), c);
    c = wmma_bf16(a1, loadB(attnW2T, nt, 1, lane), c);
    int n = nt * 16 + M;
    float bias = attnb2[n];
    float* apnt = alpha + (size_t)(ebase + 8 * half) * 64 + n;
    if (full) {
#pragma unroll
      for (int v = 0; v < 8; ++v)
        __builtin_nontemporal_store(c[v] + bias, apnt + v * 64);
    } else {
#pragma unroll
      for (int v = 0; v < 8; ++v) {
        int e = ebase + v + 8 * half;
        if (e < E) alpha[e * 64 + n] = c[v] + bias;
      }
    }
  }
}

// ---------------------------------------------------------------------------
// Kernels 3-5: segment softmax + weighted scatter-add (L2 atomics).
// Edge streams use NT loads so the 192MB L2 keeps m/s/out/x_lin resident.
// ---------------------------------------------------------------------------
__global__ __launch_bounds__(256) void k_seg_max(
    const float* __restrict__ alpha, const int* __restrict__ sidx,
    unsigned int* __restrict__ m_enc, int EC) {
  int idx = blockIdx.x * 256 + threadIdx.x;
  if (idx >= EC) return;
  int e = idx >> 6, c = idx & 63;
  float a = __builtin_nontemporal_load(alpha + idx);
  atomicMax(&m_enc[sidx[e] * 64 + c], fenc(a));
}

__global__ __launch_bounds__(256) void k_seg_expsum(
    float* __restrict__ alpha, const int* __restrict__ sidx,
    const unsigned int* __restrict__ m_enc, float* __restrict__ s, int EC) {
  int idx = blockIdx.x * 256 + threadIdx.x;
  if (idx >= EC) return;
  int e = idx >> 6, c = idx & 63;
  int g = sidx[e];
  float a = __builtin_nontemporal_load(alpha + idx);
  float ev = __expf(a - fdec(m_enc[g * 64 + c]));
  __builtin_nontemporal_store(ev, alpha + idx);   // overwrite in place with e
  atomicAdd(&s[g * 64 + c], ev);
}

__global__ __launch_bounds__(256) void k_final(
    const float* __restrict__ eexp, const int* __restrict__ sidx, const float* __restrict__ sv,
    const float* __restrict__ s, const float* __restrict__ xlin, const float* __restrict__ delta,
    const int* __restrict__ src, const int* __restrict__ dst,
    float* __restrict__ out, int EC) {
  int idx = blockIdx.x * 256 + threadIdx.x;
  if (idx >= EC) return;
  int e = idx >> 6, c = idx & 63;
  int g = sidx[e];
  float ev = __builtin_nontemporal_load(eexp + idx);
  float dl = __builtin_nontemporal_load(delta + idx);
  float attn = ev / (s[g * 64 + c] + 1e-16f) * sv[e];
  float msg  = attn * (xlin[src[e] * 64 + c] + dl);
  atomicAdd(&out[dst[e] * 64 + c], msg);
}

// ---------------------------------------------------------------------------
extern "C" void kernel_launch(void* const* d_in, const int* in_sizes, int n_in,
                              void* d_out, int out_size, void* d_ws, size_t ws_size,
                              hipStream_t stream) {
  const float* x      = (const float*)d_in[0];
  const float* pos    = (const float*)d_in[1];
  const int*   ei     = (const int*)d_in[2];
  const int*   sidx   = (const int*)d_in[3];
  const float* sv     = (const float*)d_in[4];
  const float* Wlin   = (const float*)d_in[5];
  const float* Wsrc   = (const float*)d_in[6];
  const float* Wdst   = (const float*)d_in[7];
  const float* posW1  = (const float*)d_in[8];
  const float* posb1  = (const float*)d_in[9];
  const float* posW2  = (const float*)d_in[10];
  const float* posb2  = (const float*)d_in[11];
  const float* attnW1 = (const float*)d_in[12];
  const float* attnb1 = (const float*)d_in[13];
  const float* attnW2 = (const float*)d_in[14];
  const float* attnb2 = (const float*)d_in[15];

  const int N  = in_sizes[0] / 64;
  const int E  = in_sizes[3];
  const int NC = N * 64;
  const int EC = E * 64;
  const int* src = ei;
  const int* dst = ei + E;

  // workspace carve-up
  char* ws = (char*)d_ws;
  size_t off = 0;
  auto alloc = [&](size_t bytes) -> void* {
    void* p = ws + off;
    off = (off + bytes + 255) & ~(size_t)255;
    return p;
  };
  unsigned short* xb    = (unsigned short*)alloc((size_t)NC * 2);
  unsigned short* wbuf  = (unsigned short*)alloc(6 * 4096 * 2);
  float*          xlin  = (float*)alloc((size_t)NC * 4);
  float*          asrc  = (float*)alloc((size_t)NC * 4);
  float*          adst  = (float*)alloc((size_t)NC * 4);
  float*          dlt   = (float*)alloc((size_t)EC * 4);
  float*          alp   = (float*)alloc((size_t)EC * 4);
  unsigned int*   m_enc = (unsigned int*)alloc((size_t)NC * 4);
  float*          ssum  = (float*)alloc((size_t)NC * 4);
  float*          out   = (float*)d_out;

  // 0: pack / init
  {
    int blocks = (NC + 255) / 256;
    k_init<<<blocks, 256, 0, stream>>>(x, Wlin, Wsrc, Wdst, posW2, attnW1, attnW2,
                                       xb, wbuf, m_enc, ssum, out, NC);
  }
  // 1: node GEMMs
  {
    int units = ((N + 15) / 16) * 3;
    int blocks = (units + 7) / 8;
    k_node_gemm<<<blocks, 256, 0, stream>>>(xb, wbuf, xlin, asrc, adst, N);
  }
  // 2: edge MLPs
  {
    int tiles = (E + 15) / 16;
    int blocks = (tiles + EW - 1) / EW;
    k_edge_mlp<<<blocks, 256, 0, stream>>>(pos, src, dst, asrc, adst, wbuf,
                                           posW1, posb1, posb2, attnb1, attnb2,
                                           dlt, alp, E);
  }
  // 3-5: segment softmax + scatter
  {
    int blocks = (EC + 255) / 256;
    k_seg_max<<<blocks, 256, 0, stream>>>(alp, sidx, m_enc, EC);
    k_seg_expsum<<<blocks, 256, 0, stream>>>(alp, sidx, m_enc, ssum, EC);
    k_final<<<blocks, 256, 0, stream>>>(alp, sidx, sv, ssum, xlin, dlt, src, dst, out, EC);
  }
}
